// RWKV7CrossAttention_21225728377515
// MI455X (gfx1250) — matrix operand order repaired and checked by reading, not compiled, not run on hardware
//
#include <hip/hip_runtime.h>
#include <hip/hip_bf16.h>
#include <cstdint>
#include <cstddef>

// ---------------------------------------------------------------------------
// RWKV7 cross-attention forward for MI455X (gfx1250, wave32, WMMA).
// GEMMs: bf16 v_wmma_f32_16x16x32_bf16 with f32 accumulate.
// Scan:  register-resident 64x64 state per (b,h), LDS-broadcast operands.
// ---------------------------------------------------------------------------

typedef __attribute__((ext_vector_type(16))) __bf16 v16bf;
typedef __attribute__((ext_vector_type(8)))  __bf16 v8bf;
typedef __attribute__((ext_vector_type(8)))  float  v8f;

#define B_    2
#define T_    2048
#define H_    1024
#define NH_   16
#define DK_   64
#define VD_   1024
#define DV_   64
#define LCTX_ 512
#define CD_   768
#define BT_   (B_*T_)            // 4096 rows
#define EPS_  (64.0e-5f)         // DK * 1e-5
#define LOGW_SCALE_ (-0.6065306597126334f)

__device__ __forceinline__ float sigmoidf_(float x) { return 1.0f / (1.0f + __expf(-x)); }

// ----- WMMA fragment loaders ------------------------------------------------
// A (16x32 bf16, row-major source): lanes 0-15 rows m, K = 16*half..+15
// (contiguous, matches packed B)  -> two 16B loads per lane.
__device__ __forceinline__ v16bf load_a_frag(const __bf16* A, int lda,
                                             int mBase, int kBase, int lane) {
  const int m = mBase + (lane & 15);
  const int half = lane >> 4;
  const __bf16* p = A + (size_t)m * lda + kBase + 16 * half;
  v8bf lo = *(const v8bf*)(p);
  v8bf hi = *(const v8bf*)(p + 8);
  v16bf r;
#pragma unroll
  for (int i = 0; i < 8; ++i) { r[i] = lo[i]; r[i + 8] = hi[i]; }
  return r;
}

// B packed layout: tile (nt,kt) of 32(K)x16(N); lane owns 16 contiguous bf16
// = B[kt*32 + 16*(lane>>4) + e][nt*16 + (lane&15)], e=0..15.
__device__ __forceinline__ v16bf load_b_frag(const __bf16* Bp, int KT,
                                             int nt, int kt, int lane) {
  const __bf16* p = Bp + ((size_t)(nt * KT + kt) * 32 + lane) * 16;
  v8bf lo = *(const v8bf*)(p);
  v8bf hi = *(const v8bf*)(p + 8);
  v16bf r;
#pragma unroll
  for (int i = 0; i < 8; ++i) { r[i] = lo[i]; r[i + 8] = hi[i]; }
  return r;
}

// ----- pack kernels ---------------------------------------------------------
__global__ void pack_cast_bf16(const float* __restrict__ src,
                               __bf16* __restrict__ dst, int n) {
  int i = blockIdx.x * blockDim.x + threadIdx.x;
  if (i < n) dst[i] = (__bf16)src[i];
}

__global__ void pack_b_wmma(const float* __restrict__ src, __bf16* __restrict__ dst,
                            int K, int N, int Kp, int Np64) {
  int total = Np64 * Kp;
  int idx = blockIdx.x * blockDim.x + threadIdx.x;
  if (idx >= total) return;
  int e = idx & 15;
  int lane = (idx >> 4) & 31;
  int tile = idx >> 9;
  int KT = Kp >> 5;
  int kt = tile % KT;
  int nt = tile / KT;
  int k = kt * 32 + 16 * (lane >> 4) + e;
  int n = nt * 16 + (lane & 15);
  float v = (k < K && n < N) ? src[(size_t)k * N + n] : 0.0f;
  dst[idx] = (__bf16)v;
}

// ----- WMMA GEMM with fused epilogues --------------------------------------
enum { EPI_F32 = 0, EPI_BF16 = 1, EPI_TANH_BF16 = 2, EPI_SIG_BF16 = 3,
       EPI_WDECAY = 4, EPI_SIGBIAS = 5, EPI_VLERP = 6 };

template <int EPI>
__global__ __launch_bounds__(128)
void gemm_bf16_wmma(const __bf16* __restrict__ A, int lda,
                    const __bf16* __restrict__ Bp, int Kp,
                    float* __restrict__ C, __bf16* __restrict__ Cb, int ldc,
                    int M, int Nstore,
                    const float* __restrict__ bias,
                    const float* __restrict__ aux) {
  const int lane  = threadIdx.x & 31;
  const int wave  = threadIdx.x >> 5;
  const int half  = lane >> 4;
  const int KT    = Kp >> 5;
  const int waveM = blockIdx.y * 64 + (wave >> 1) * 32;
  const int waveN = blockIdx.x * 64 + (wave & 1) * 32;
  const int nt0   = waveN >> 4;

  v8f z = {0.f, 0.f, 0.f, 0.f, 0.f, 0.f, 0.f, 0.f};
  v8f acc[2][2];
  acc[0][0] = z; acc[0][1] = z; acc[1][0] = z; acc[1][1] = z;

  for (int kt = 0; kt < KT; ++kt) {
    v16bf a0 = load_a_frag(A, lda, waveM,      kt * 32, lane);
    v16bf a1 = load_a_frag(A, lda, waveM + 16, kt * 32, lane);
    v16bf b0 = load_b_frag(Bp, KT, nt0,     kt, lane);
    v16bf b1 = load_b_frag(Bp, KT, nt0 + 1, kt, lane);
    acc[0][0] = __builtin_amdgcn_wmma_f32_16x16x32_bf16(false, a0, false, b0, (short)0, acc[0][0], false, false);
    acc[0][1] = __builtin_amdgcn_wmma_f32_16x16x32_bf16(false, a0, false, b1, (short)0, acc[0][1], false, false);
    acc[1][0] = __builtin_amdgcn_wmma_f32_16x16x32_bf16(false, a1, false, b0, (short)0, acc[1][0], false, false);
    acc[1][1] = __builtin_amdgcn_wmma_f32_16x16x32_bf16(false, a1, false, b1, (short)0, acc[1][1], false, false);
  }

#pragma unroll
  for (int i = 0; i < 2; ++i) {
#pragma unroll
    for (int j = 0; j < 2; ++j) {
      int n = waveN + j * 16 + (lane & 15);
      if (n >= Nstore) continue;
      float bv = 0.0f;
      if (EPI == EPI_WDECAY || EPI == EPI_SIGBIAS || EPI == EPI_VLERP) bv = bias[n];
#pragma unroll
      for (int r = 0; r < 8; ++r) {
        int m = waveM + i * 16 + 8 * half + r;   // ISA C/D layout (7.12.2)
        if (m >= M) continue;
        float x = acc[i][j][r];
        size_t idx = (size_t)m * ldc + n;
        if (EPI == EPI_F32)            C[idx]  = x;
        else if (EPI == EPI_BF16)      Cb[idx] = (__bf16)x;
        else if (EPI == EPI_TANH_BF16) Cb[idx] = (__bf16)tanhf(x);
        else if (EPI == EPI_SIG_BF16)  Cb[idx] = (__bf16)sigmoidf_(x);
        else if (EPI == EPI_WDECAY)    C[idx]  = LOGW_SCALE_ * sigmoidf_(x + bv);
        else if (EPI == EPI_SIGBIAS)   C[idx]  = sigmoidf_(x + bv);
        else if (EPI == EPI_VLERP) {
          float v0 = C[idx];
          float vf = aux[idx];
          C[idx] = v0 + sigmoidf_(x + bv) * (vf - v0);
        }
      }
    }
  }
}

// ----- fused resample + token-shift + mix ----------------------------------
__global__ void mix_kernel(const float* __restrict__ hs, const float* __restrict__ ctxp,
                           const float* __restrict__ xr_c, const float* __restrict__ xw_c,
                           const float* __restrict__ xk_c, const float* __restrict__ xv_c,
                           const float* __restrict__ xa_c, const float* __restrict__ xg_c,
                           __bf16* __restrict__ xr, __bf16* __restrict__ xw,
                           __bf16* __restrict__ xk, __bf16* __restrict__ xv,
                           __bf16* __restrict__ xa, __bf16* __restrict__ xg) {
  int idx = blockIdx.x * blockDim.x + threadIdx.x;
  if (idx >= B_ * T_ * H_) return;
  int h = idx % H_;
  int t = (idx / H_) % T_;
  int b = idx / (H_ * T_);
  const float sc = (float)(LCTX_ - 1) / (float)(T_ - 1);
  const float* cb = ctxp + (size_t)b * LCTX_ * H_ + h;

  float pos = t * sc;
  int lo = (int)pos; int hi = lo + 1; if (hi > LCTX_ - 1) hi = LCTX_ - 1;
  float f = pos - (float)lo;
  float ct = cb[(size_t)lo * H_] * (1.f - f) + cb[(size_t)hi * H_] * f;

  float cp = 0.f;
  if (t > 0) {
    float p2 = (t - 1) * sc;
    int l2 = (int)p2; int h2 = l2 + 1; if (h2 > LCTX_ - 1) h2 = LCTX_ - 1;
    float f2 = p2 - (float)l2;
    cp = cb[(size_t)l2 * H_] * (1.f - f2) + cb[(size_t)h2 * H_] * f2;
  }
  float ht = hs[idx];
  float hp = (t > 0) ? hs[idx - H_] : 0.f;
  float dx = hp - ht, dc = cp - ct;
  xr[idx] = (__bf16)(ht + dx * xr_c[h]);
  xw[idx] = (__bf16)(ht + dx * xw_c[h]);
  xa[idx] = (__bf16)(ht + dx * xa_c[h]);
  xg[idx] = (__bf16)(ht + dx * xg_c[h]);
  xk[idx] = (__bf16)(ct + dc * xk_c[h]);
  xv[idx] = (__bf16)(ct + dc * xv_c[h]);
}

// ----- per-head key prep: kk-norm, ah/bh, k update, r*k*r_k dot ------------
__global__ __launch_bounds__(64)
void prep_kernel(float* __restrict__ k, const float* __restrict__ a,
                 const float* __restrict__ r,
                 const float* __restrict__ k_k, const float* __restrict__ k_a,
                 const float* __restrict__ r_k,
                 float* __restrict__ ah, float* __restrict__ bh,
                 float* __restrict__ rk_dot) {
  __shared__ float red[64];
  int g = blockIdx.x;                 // bt*NH + h
  int h = g % NH_;
  int i = threadIdx.x;
  size_t base = (size_t)g * 64;
  int col = h * 64 + i;

  float kv  = k[base + i];
  float av  = a[base + i];
  float kkv = kv * k_k[col];
  red[i] = kkv * kkv; __syncthreads();
  for (int s = 32; s > 0; s >>= 1) { if (i < s) red[i] += red[i + s]; __syncthreads(); }
  float den = fmaxf(sqrtf(red[0]), 1e-12f);
  __syncthreads();
  float kkn = kkv / den;
  ah[base + i] = -kkn;
  bh[base + i] = kkn * av;
  float kf = kv * (1.0f + (av - 1.0f) * k_a[col]);
  k[base + i] = kf;                   // in-place final key
  red[i] = r[base + i] * kf * r_k[col]; __syncthreads();
  for (int s = 32; s > 0; s >>= 1) { if (i < s) red[i] += red[i + s]; __syncthreads(); }
  if (i == 0) rk_dot[g] = red[0];
}

// ----- RWKV7 delta-rule scan: S[64x64] in registers, 1 block per (b,h) -----
__global__ __launch_bounds__(64)
void rwkv_scan(const float* __restrict__ r, const float* __restrict__ w,
               const float* __restrict__ k, const float* __restrict__ v,
               const float* __restrict__ ah, const float* __restrict__ bh,
               float* __restrict__ o) {
  __shared__ float lw[64], lk[64], la[64], lb[64], lr[64];
  int bh_i = blockIdx.x;
  int b = bh_i / NH_;
  int h = bh_i % NH_;
  int i = threadIdx.x;                // value index (state row)
  float S[64];
#pragma unroll
  for (int j = 0; j < 64; ++j) S[j] = 0.f;

  for (int t = 0; t < T_; ++t) {
    size_t base = ((size_t)(b * T_ + t) * NH_ + h) * 64;
    lw[i] = __expf(w[base + i]);
    lk[i] = k[base + i];
    la[i] = ah[base + i];
    lb[i] = bh[base + i];
    lr[i] = r[base + i];
    float vi = v[base + i];
    __syncthreads();
    float sa = 0.f;
#pragma unroll
    for (int j = 0; j < 64; ++j) sa += S[j] * la[j];
    float oi = 0.f;
#pragma unroll
    for (int j = 0; j < 64; ++j) {
      S[j] = S[j] * lw[j] + sa * lb[j] + vi * lk[j];
      oi += S[j] * lr[j];
    }
    o[base + i] = oi;
    __syncthreads();
  }
}

// ----- GroupNorm + direct readout + gate, emit bf16 for final GEMM ---------
__global__ __launch_bounds__(64)
void out_norm_kernel(const float* __restrict__ o, const float* __restrict__ v,
                     const float* __restrict__ g, const float* __restrict__ rk_dot,
                     const float* __restrict__ gn_w, const float* __restrict__ gn_b,
                     __bf16* __restrict__ og) {
  __shared__ float red[64];
  int grp = blockIdx.x;
  int h = grp % NH_;
  int i = threadIdx.x;
  size_t base = (size_t)grp * 64;
  float x = o[base + i];
  red[i] = x; __syncthreads();
  for (int s = 32; s > 0; s >>= 1) { if (i < s) red[i] += red[i + s]; __syncthreads(); }
  float mu = red[0] * (1.0f / 64.0f);
  __syncthreads();
  float d = x - mu;
  red[i] = d * d; __syncthreads();
  for (int s = 32; s > 0; s >>= 1) { if (i < s) red[i] += red[i + s]; __syncthreads(); }
  float var = red[0] * (1.0f / 64.0f);
  int col = h * 64 + i;
  float on = d * rsqrtf(var + EPS_) * gn_w[col] + gn_b[col] + rk_dot[grp] * v[base + i];
  og[base + i] = (__bf16)(on * g[base + i]);
}

// ---------------------------------------------------------------------------
static inline int cdiv_(int a, int b) { return (a + b - 1) / b; }

static void launch_pack_b(const float* src, __bf16* dst, int K, int N, hipStream_t s) {
  int Kp = cdiv_(K, 32) * 32;
  int Np = cdiv_(N, 64) * 64;
  int total = Np * Kp;
  pack_b_wmma<<<cdiv_(total, 256), 256, 0, s>>>(src, dst, K, N, Kp, Np);
}

extern "C" void kernel_launch(void* const* d_in, const int* in_sizes, int n_in,
                              void* d_out, int out_size, void* d_ws, size_t ws_size,
                              hipStream_t stream) {
  (void)in_sizes; (void)n_in; (void)out_size; (void)ws_size;
  const float* hs     = (const float*)d_in[0];
  const float* ctx    = (const float*)d_in[1];
  const float* vfirst = (const float*)d_in[2];
  const float* x_r    = (const float*)d_in[3];
  const float* x_w    = (const float*)d_in[4];
  const float* x_k    = (const float*)d_in[5];
  const float* x_v    = (const float*)d_in[6];
  const float* x_a    = (const float*)d_in[7];
  const float* x_g    = (const float*)d_in[8];
  const float* k_k    = (const float*)d_in[9];
  const float* k_a    = (const float*)d_in[10];
  const float* r_k    = (const float*)d_in[11];
  const float* W_r    = (const float*)d_in[12];
  const float* W_k    = (const float*)d_in[13];
  const float* W_v    = (const float*)d_in[14];
  const float* W_o    = (const float*)d_in[15];
  const float* W_cond = (const float*)d_in[16];
  const float* wA     = (const float*)d_in[17];
  const float* wB     = (const float*)d_in[18];
  const float* wb     = (const float*)d_in[19];
  const float* aA     = (const float*)d_in[20];
  const float* aB     = (const float*)d_in[21];
  const float* ab     = (const float*)d_in[22];
  const float* gA     = (const float*)d_in[23];
  const float* gB     = (const float*)d_in[24];
  const float* vA     = (const float*)d_in[25];
  const float* vB     = (const float*)d_in[26];
  const float* vb     = (const float*)d_in[27];
  const float* gn_w   = (const float*)d_in[28];
  const float* gn_b   = (const float*)d_in[29];
  float* out = (float*)d_out;

  char* ws = (char*)d_ws;
  size_t off = 0;
  auto alloc = [&](size_t bytes) -> void* {
    off = (off + 255) & ~(size_t)255;
    void* p = ws + off;
    off += bytes;
    return p;
  };
  auto allocF = [&](size_t e) -> float*  { return (float*) alloc(e * 4); };
  auto allocB = [&](size_t e) -> __bf16* { return (__bf16*)alloc(e * 2); };

  // bf16 buffers
  __bf16* ctxB  = allocB((size_t)B_ * LCTX_ * CD_);       // 1024x768
  __bf16* pWcond= allocB((size_t)1024 * 768);
  __bf16* pWr   = allocB((size_t)1024 * 1024);
  __bf16* pWk   = allocB((size_t)1024 * 1024);
  __bf16* pWv   = allocB((size_t)1024 * 1024);
  __bf16* pWo   = allocB((size_t)1024 * 1024);
  __bf16* pwA   = allocB((size_t)64 * 1024);
  __bf16* pwB   = allocB((size_t)1024 * 64);
  __bf16* paA   = allocB((size_t)64 * 1024);
  __bf16* paB   = allocB((size_t)1024 * 64);
  __bf16* pgA   = allocB((size_t)128 * 1024);
  __bf16* pgB   = allocB((size_t)1024 * 128);
  __bf16* pvA   = allocB((size_t)64 * 1024);
  __bf16* pvB   = allocB((size_t)1024 * 32);
  __bf16* xr    = allocB((size_t)BT_ * H_);
  __bf16* xw    = allocB((size_t)BT_ * H_);
  __bf16* xk    = allocB((size_t)BT_ * H_);
  __bf16* xv    = allocB((size_t)BT_ * H_);
  __bf16* xa    = allocB((size_t)BT_ * H_);
  __bf16* xg    = allocB((size_t)BT_ * H_);
  __bf16* t1    = allocB((size_t)BT_ * 64);
  __bf16* t2    = allocB((size_t)BT_ * 64);   // N=16 padded to 64
  __bf16* t3    = allocB((size_t)BT_ * 64);
  __bf16* t4    = allocB((size_t)BT_ * 128);
  __bf16* og    = allocB((size_t)BT_ * VD_);
  // f32 buffers
  float* ctxp = allocF((size_t)B_ * LCTX_ * H_);
  float* rbuf = allocF((size_t)BT_ * H_);
  float* wbuf = allocF((size_t)BT_ * H_);
  float* kbuf = allocF((size_t)BT_ * H_);
  float* vbuf = allocF((size_t)BT_ * VD_);
  float* abuf = allocF((size_t)BT_ * H_);
  float* gbuf = allocF((size_t)BT_ * VD_);
  float* ahb  = allocF((size_t)BT_ * H_);
  float* bhb  = allocF((size_t)BT_ * H_);
  float* obuf = allocF((size_t)BT_ * VD_);
  float* rk   = allocF((size_t)BT_ * NH_);

  // ---- packs ----
  {
    int n = B_ * LCTX_ * CD_;
    pack_cast_bf16<<<cdiv_(n, 256), 256, 0, stream>>>(ctx, ctxB, n);
  }
  launch_pack_b(W_cond, pWcond, CD_, H_,   stream);
  launch_pack_b(W_r,    pWr,    H_,  H_,   stream);
  launch_pack_b(W_k,    pWk,    H_,  H_,   stream);
  launch_pack_b(W_v,    pWv,    H_,  VD_,  stream);
  launch_pack_b(W_o,    pWo,    VD_, H_,   stream);
  launch_pack_b(wA,     pwA,    H_,  64,   stream);
  launch_pack_b(wB,     pwB,    64,  H_,   stream);
  launch_pack_b(aA,     paA,    H_,  64,   stream);
  launch_pack_b(aB,     paB,    64,  H_,   stream);
  launch_pack_b(gA,     pgA,    H_,  128,  stream);
  launch_pack_b(gB,     pgB,    128, VD_,  stream);
  launch_pack_b(vA,     pvA,    H_,  16,   stream);
  launch_pack_b(vB,     pvB,    16,  VD_,  stream);

  dim3 blk(128);
  // ctx projection: [1024,768] x [768,1024] -> f32
  gemm_bf16_wmma<EPI_F32><<<dim3(cdiv_(1024, 64), cdiv_(1024, 64)), blk, 0, stream>>>(
      ctxB, CD_, pWcond, 768, ctxp, nullptr, H_, 1024, 1024, nullptr, nullptr);

  // resample + token shift + mix
  {
    int n = B_ * T_ * H_;
    mix_kernel<<<cdiv_(n, 256), 256, 0, stream>>>(hs, ctxp, x_r, x_w, x_k, x_v, x_a, x_g,
                                                  xr, xw, xk, xv, xa, xg);
  }

  dim3 gBig(cdiv_(1024, 64), cdiv_(BT_, 64));
  // r = xr @ W_r
  gemm_bf16_wmma<EPI_F32><<<gBig, blk, 0, stream>>>(xr, H_, pWr, H_, rbuf, nullptr, H_, BT_, H_, nullptr, nullptr);
  // t1 = tanh(xw @ wA)
  gemm_bf16_wmma<EPI_TANH_BF16><<<dim3(1, cdiv_(BT_, 64)), blk, 0, stream>>>(xw, H_, pwA, H_, nullptr, t1, 64, BT_, 64, nullptr, nullptr);
  // w = LOGW_SCALE * sigmoid(t1 @ wB + wb)
  gemm_bf16_wmma<EPI_WDECAY><<<gBig, blk, 0, stream>>>(t1, 64, pwB, 64, wbuf, nullptr, H_, BT_, H_, wb, nullptr);
  // k = xk @ W_k
  gemm_bf16_wmma<EPI_F32><<<gBig, blk, 0, stream>>>(xk, H_, pWk, H_, kbuf, nullptr, H_, BT_, H_, nullptr, nullptr);
  // v0 = xv @ W_v
  gemm_bf16_wmma<EPI_F32><<<gBig, blk, 0, stream>>>(xv, H_, pWv, H_, vbuf, nullptr, VD_, BT_, VD_, nullptr, nullptr);
  // t2 = xv @ vA   (N=16 padded to 64, zero cols)
  gemm_bf16_wmma<EPI_BF16><<<dim3(1, cdiv_(BT_, 64)), blk, 0, stream>>>(xv, H_, pvA, H_, nullptr, t2, 64, BT_, 64, nullptr, nullptr);
  // v = v0 + sigmoid(t2 @ vB + vb) * (v_first - v0)   (in place on vbuf)
  gemm_bf16_wmma<EPI_VLERP><<<gBig, blk, 0, stream>>>(t2, 64, pvB, 32, vbuf, nullptr, VD_, BT_, VD_, vb, vfirst);
  // t3 = xa @ aA
  gemm_bf16_wmma<EPI_BF16><<<dim3(1, cdiv_(BT_, 64)), blk, 0, stream>>>(xa, H_, paA, H_, nullptr, t3, 64, BT_, 64, nullptr, nullptr);
  // a = sigmoid(t3 @ aB + ab)
  gemm_bf16_wmma<EPI_SIGBIAS><<<gBig, blk, 0, stream>>>(t3, 64, paB, 64, abuf, nullptr, H_, BT_, H_, ab, nullptr);
  // t4 = sigmoid(xg @ gA)
  gemm_bf16_wmma<EPI_SIG_BF16><<<dim3(cdiv_(128, 64), cdiv_(BT_, 64)), blk, 0, stream>>>(xg, H_, pgA, H_, nullptr, t4, 128, BT_, 128, nullptr, nullptr);
  // g = t4 @ gB
  gemm_bf16_wmma<EPI_F32><<<gBig, blk, 0, stream>>>(t4, 128, pgB, 128, gbuf, nullptr, VD_, BT_, VD_, nullptr, nullptr);

  // per-head prep (norm(kk), ah, bh, k-final, rk_dot)
  prep_kernel<<<BT_ * NH_, 64, 0, stream>>>(kbuf, abuf, rbuf, k_k, k_a, r_k, ahb, bhb, rk);

  // delta-rule scan, 32 independent chains
  rwkv_scan<<<B_ * NH_, 64, 0, stream>>>(rbuf, wbuf, kbuf, vbuf, ahb, bhb, obuf);

  // GroupNorm + readout + gate -> bf16
  out_norm_kernel<<<BT_ * NH_, 64, 0, stream>>>(obuf, vbuf, gbuf, rk, gn_w, gn_b, og);

  // final: out = og @ W_o  (f32 output)
  gemm_bf16_wmma<EPI_F32><<<gBig, blk, 0, stream>>>(og, VD_, pWo, VD_, out, nullptr, H_, BT_, H_, nullptr, nullptr);
}
